// DecoderModel_20444044329725
// MI455X (gfx1250) — matrix operand order repaired and checked by reading, not compiled, not run on hardware
//
#include <hip/hip_runtime.h>
#include <math.h>

// ---------------- constants (match reference) ----------------
#define VOCAB 32000
#define EMB   512
#define HID   256
#define LAYERS 4
#define BATCH 64
#define SRC   128
#define TSTEPS 32
#define GATES (4 * HID)   // 1024

#define LDS_STRIDE 520    // 512 + 8 shorts pad: row stride 1040B = 260 dwords == 4 mod 64 banks

// ---------------- WMMA types ----------------
typedef __attribute__((ext_vector_type(16))) __bf16 v16bf;
typedef __attribute__((ext_vector_type(8)))  float  v8f;

union ABPack {
    uint4 q[2];
    v16bf v;
};

__device__ inline unsigned short f2bf(float f) {
    unsigned int x = __float_as_uint(f);
    unsigned int r = x + 0x7FFFu + ((x >> 16) & 1u);   // RNE
    return (unsigned short)(r >> 16);
}

__device__ inline float sigm(float x) { return 1.0f / (1.0f + expf(-x)); }

// ---------------- f32 -> bf16 weight conversion ----------------
__global__ void f32_to_bf16_kernel(const float* __restrict__ src,
                                   unsigned short* __restrict__ dst, int n) {
    int i = blockIdx.x * 256 + threadIdx.x;
    if (i < n) dst[i] = f2bf(src[i]);
}

// ---------------- state init ----------------
__global__ void init_state_kernel(float* h, float* c, unsigned short* hbf,
                                  int* tok, const int* start_code) {
    int i = blockIdx.x * 256 + threadIdx.x;
    const int n = LAYERS * BATCH * HID;
    if (i < n) { h[i] = 0.0f; c[i] = 0.0f; hbf[i] = 0; }
    if (i < BATCH) tok[i] = start_code[0];
}

// ---------------- embedding gather (fp32 -> bf16 rows) ----------------
__global__ void embed_gather_kernel(const float* __restrict__ emb,
                                    const int* __restrict__ tok,
                                    unsigned short* __restrict__ x_bf) {
    int i = blockIdx.x * 256 + threadIdx.x;           // 0 .. 64*512-1
    int b = i >> 9;
    int k = i & (EMB - 1);
    x_bf[i] = f2bf(emb[(size_t)tok[b] * EMB + k]);
}

// ---------------- bf16 WMMA K-loop, 4 M-tiles per wave (B reused 4x) ----------------
// A: [64, K] row-major bf16 ; W: [N, K] row-major bf16 (acts as B = W^T)
__device__ inline void wmma_strip_acc(const unsigned short* __restrict__ A,
                                      const unsigned short* __restrict__ W,
                                      int K, int mrow, int n, int hi, v8f acc[4]) {
    for (int kk = 0; kk < K; kk += 32) {
        ABPack ub;
        const unsigned short* bp = W + (size_t)n * K + kk + hi * 8;
        ub.q[0] = *(const uint4*)(bp);        // K = kk+hi*8     .. +7
        ub.q[1] = *(const uint4*)(bp + 16);   // K = kk+hi*8+16  .. +23
#pragma unroll
        for (int mt = 0; mt < 4; ++mt) {
            ABPack ua;
            const unsigned short* ap = A + (size_t)(mt * 16 + mrow) * K + kk + hi * 8;
            ua.q[0] = *(const uint4*)(ap);        // K = kk+hi*8     .. +7
            ua.q[1] = *(const uint4*)(ap + 16);   // K = kk+hi*8+16  .. +23
            acc[mt] = __builtin_amdgcn_wmma_f32_16x16x32_bf16(
                false, ua.v, false, ub.v, (short)0, acc[mt], false, false);
        }
    }
}

// ---------------- LSTM gate GEMM: gates[64,1024] = A1@W1^T + A2@W2^T ----------------
// 64 waves total: nt = wave (64 N tiles of 1024); each wave owns full M=64 strip
__global__ __launch_bounds__(256)
void lstm_gates_wmma_kernel(const unsigned short* __restrict__ A1, int K1,
                            const unsigned short* __restrict__ A2,           // [64, HID]
                            const unsigned short* __restrict__ W1,           // [1024, K1]
                            const unsigned short* __restrict__ W2,           // [1024, HID]
                            float* __restrict__ gates) {
    int nt   = blockIdx.x * 8 + (threadIdx.x >> 5);   // 0..63
    int lane = threadIdx.x & 31;
    int hi   = lane >> 4;
    int mrow = lane & 15;
    int n = nt * 16 + mrow;

    v8f acc[4] = {v8f{}, v8f{}, v8f{}, v8f{}};
    wmma_strip_acc(A1, W1, K1,  mrow, n, hi, acc);
    wmma_strip_acc(A2, W2, HID, mrow, n, hi, acc);

    int col = nt * 16 + (lane & 15);
#pragma unroll
    for (int mt = 0; mt < 4; ++mt) {
        int m0 = mt * 16 + hi * 8;
#pragma unroll
        for (int r = 0; r < 8; ++r)
            gates[(size_t)(m0 + r) * GATES + col] = acc[mt][r];
    }
}

// ---------------- LSTM pointwise: i,f,g,o -> h,c (per layer) ----------------
__global__ void lstm_pointwise_kernel(const float* __restrict__ gates,
                                      const float* __restrict__ bias,   // [1024]
                                      float* __restrict__ h_f32,        // [64,256] this layer
                                      float* __restrict__ c_f32,
                                      unsigned short* __restrict__ h_bf) {
    int idx = blockIdx.x * 256 + threadIdx.x;   // 0 .. 64*256-1
    int b = idx >> 8;
    int j = idx & (HID - 1);
    const float* g = gates + (size_t)b * GATES;
    float i_ = sigm(g[j]           + bias[j]);
    float f_ = sigm(g[HID + j]     + bias[HID + j]);
    float g_ = tanhf(g[2 * HID + j] + bias[2 * HID + j]);
    float o_ = sigm(g[3 * HID + j] + bias[3 * HID + j]);
    float c  = f_ * c_f32[idx] + i_ * g_;
    float h  = o_ * tanhf(c);
    c_f32[idx] = c;
    h_f32[idx] = h;
    h_bf[idx]  = f2bf(h);
}

// ---------------- attention scores: scores[s,b] = h_top[b] . enc_h[s,b,:] ----------------
__global__ void attn_scores_kernel(const float* __restrict__ enc_h,
                                   const float* __restrict__ h_top,
                                   float* __restrict__ scores) {
    int s = blockIdx.x;        // 0..127
    int b = threadIdx.x;       // 0..63
    const float* e = enc_h + ((size_t)s * BATCH + b) * HID;
    const float* h = h_top + (size_t)b * HID;
    float acc = 0.0f;
#pragma unroll 4
    for (int d = 0; d < HID; ++d) acc += h[d] * e[d];
    scores[s * BATCH + b] = acc;
}

// ---------------- softmax over S + context + build h_c (bf16 [64,512]) ----------------
__global__ void attn_ctx_kernel(const float* __restrict__ enc_h,
                                const float* __restrict__ h_top,
                                const float* __restrict__ scores,
                                unsigned short* __restrict__ h_c_bf) {
    int b = blockIdx.x;        // 0..63
    int t = threadIdx.x;       // 0..127
    __shared__ float a[SRC];
    __shared__ float red[SRC];

    float v = scores[t * BATCH + b];
    red[t] = v;
    __syncthreads();
    for (int off = 64; off > 0; off >>= 1) {
        if (t < off) red[t] = fmaxf(red[t], red[t + off]);
        __syncthreads();
    }
    float mx = red[0];
    __syncthreads();
    float e = expf(v - mx);
    red[t] = e;
    __syncthreads();
    for (int off = 64; off > 0; off >>= 1) {
        if (t < off) red[t] += red[t + off];
        __syncthreads();
    }
    float inv = 1.0f / red[0];
    __syncthreads();
    a[t] = e * inv;
    __syncthreads();

#pragma unroll
    for (int dd = 0; dd < 2; ++dd) {
        int d = t + dd * SRC;                 // 0..255
        float acc = 0.0f;
        for (int s = 0; s < SRC; ++s)
            acc += a[s] * enc_h[((size_t)s * BATCH + b) * HID + d];
        h_c_bf[(size_t)b * (2 * HID) + HID + d] = f2bf(acc);                  // ctx half
        h_c_bf[(size_t)b * (2 * HID) + d] = f2bf(h_top[(size_t)b * HID + d]); // h_t half
    }
}

// ---------------- FC: sigmoid(h_c @ w_fc^T + b_fc) -> d_out row ----------------
// 2000 waves: nt = wave (0..1999); each wave owns the full M=64 strip.
// A (h_c, 64x512 bf16) staged once per block into LDS via async loads.
__global__ __launch_bounds__(256)
void fc_wmma_kernel(const unsigned short* __restrict__ A,   // h_c_bf [64,512]
                    const unsigned short* __restrict__ W,   // w_fc_bf [32000,512]
                    const float* __restrict__ bias,         // [32000]
                    float* __restrict__ out) {              // [64,32000]
    __shared__ unsigned short sA[BATCH * LDS_STRIDE];       // 66,560 B

    // ---- stage A: 4096 16B chunks, 16 per thread, ASYNCcnt-tracked ----
    {
        int tid = threadIdx.x;
#pragma unroll
        for (int i = 0; i < 16; ++i) {
            int c = (tid << 4) + i;                  // 0..4095
            int m = c >> 6;                          // row 0..63
            int j = c & 63;                          // 8-short chunk in row
            unsigned lds_b = (unsigned)((m * LDS_STRIDE + j * 8) * 2);  // byte offset in LDS alloc
            unsigned long long ga =
                (unsigned long long)(uintptr_t)(A + (size_t)m * 512 + j * 8);
            asm volatile("global_load_async_to_lds_b128 %0, %1, off"
                         :: "v"(lds_b), "v"(ga) : "memory");
        }
        asm volatile("s_wait_asynccnt 0x0" ::: "memory");
    }
    __syncthreads();

    int nt   = blockIdx.x * 8 + (threadIdx.x >> 5);   // 0..1999
    int lane = threadIdx.x & 31;
    int hi   = lane >> 4;
    int mrow = lane & 15;
    int n = nt * 16 + mrow;
    const int K = 2 * HID;      // 512

    v8f acc[4] = {v8f{}, v8f{}, v8f{}, v8f{}};
    for (int kk = 0; kk < K; kk += 32) {
        ABPack ub;
        const unsigned short* bp = W + (size_t)n * K + kk + hi * 8;
        __builtin_prefetch(bp + 32, 0, 1);   // next K-chunk -> global_prefetch
        ub.q[0] = *(const uint4*)(bp);
        ub.q[1] = *(const uint4*)(bp + 16);
#pragma unroll
        for (int mt = 0; mt < 4; ++mt) {
            ABPack ua;
            const unsigned short* ap = &sA[(mt * 16 + mrow) * LDS_STRIDE + kk + hi * 8];
            ua.q[0] = *(const uint4*)(ap);        // ds_load_b128
            ua.q[1] = *(const uint4*)(ap + 16);
            acc[mt] = __builtin_amdgcn_wmma_f32_16x16x32_bf16(
                false, ua.v, false, ub.v, (short)0, acc[mt], false, false);
        }
    }

    int col = nt * 16 + (lane & 15);
    float bb = bias[col];
#pragma unroll
    for (int mt = 0; mt < 4; ++mt) {
        int m0 = mt * 16 + hi * 8;
#pragma unroll
        for (int r = 0; r < 8; ++r)
            out[(size_t)(m0 + r) * VOCAB + col] = sigm(acc[mt][r] + bb);
    }
}

// ---------------- vocab softmax (in place) + argmax -> next token ----------------
__global__ void softmax_argmax_kernel(float* __restrict__ out_row, int* __restrict__ tok) {
    int b = blockIdx.x;          // 0..63
    int t = threadIdx.x;         // 0..255
    float* row = out_row + (size_t)b * VOCAB;
    __shared__ float rmax[256];
    __shared__ int   ridx[256];
    __shared__ float rsum[256];

    float mx = -INFINITY; int mi = 0;
    for (int v = t; v < VOCAB; v += 256) {
        float x = row[v];
        if (x > mx) { mx = x; mi = v; }
    }
    rmax[t] = mx; ridx[t] = mi;
    __syncthreads();
    for (int off = 128; off > 0; off >>= 1) {
        if (t < off) {
            if (rmax[t + off] > rmax[t] ||
                (rmax[t + off] == rmax[t] && ridx[t + off] < ridx[t])) {
                rmax[t] = rmax[t + off]; ridx[t] = ridx[t + off];
            }
        }
        __syncthreads();
    }
    float gmax = rmax[0]; int gidx = ridx[0];
    __syncthreads();

    float s = 0.0f;
    for (int v = t; v < VOCAB; v += 256) s += expf(row[v] - gmax);
    rsum[t] = s;
    __syncthreads();
    for (int off = 128; off > 0; off >>= 1) {
        if (t < off) rsum[t] += rsum[t + off];
        __syncthreads();
    }
    float inv = 1.0f / rsum[0];

    for (int v = t; v < VOCAB; v += 256) row[v] = expf(row[v] - gmax) * inv;
    if (t == 0) tok[b] = gidx;
}

// ---------------- host-side orchestration ----------------
extern "C" void kernel_launch(void* const* d_in, const int* in_sizes, int n_in,
                              void* d_out, int out_size, void* d_ws, size_t ws_size,
                              hipStream_t stream) {
    (void)in_sizes; (void)n_in; (void)out_size; (void)ws_size;
    const float* enc_h      = (const float*)d_in[0];
    const float* emb        = (const float*)d_in[1];
    const float* w_ih_l0    = (const float*)d_in[2];
    const float* w_hh_l0    = (const float*)d_in[3];
    const float* b_l0       = (const float*)d_in[4];
    const float* w_ih_rest  = (const float*)d_in[5];
    const float* w_hh_rest  = (const float*)d_in[6];
    const float* b_rest     = (const float*)d_in[7];
    const float* w_fc       = (const float*)d_in[8];
    const float* b_fc       = (const float*)d_in[9];
    // d_in[10] = max_sentence_len (device scalar; fixed TSTEPS at compile time)
    const int*   start_code = (const int*)d_in[11];
    float* out = (float*)d_out;

    // workspace carve-up (256B aligned)
    char* ws = (char*)d_ws;
    size_t off = 0;
    auto carve = [&](size_t bytes) -> char* {
        char* p = ws + off;
        off += (bytes + 255) & ~(size_t)255;
        return p;
    };
    unsigned short* w_fc_bf     = (unsigned short*)carve((size_t)VOCAB * 2 * HID * 2);
    unsigned short* w_ih_l0_bf  = (unsigned short*)carve((size_t)GATES * EMB * 2);
    unsigned short* w_hh_l0_bf  = (unsigned short*)carve((size_t)GATES * HID * 2);
    unsigned short* w_ih_r_bf   = (unsigned short*)carve((size_t)(LAYERS - 1) * GATES * HID * 2);
    unsigned short* w_hh_r_bf   = (unsigned short*)carve((size_t)(LAYERS - 1) * GATES * HID * 2);
    unsigned short* x_bf        = (unsigned short*)carve((size_t)BATCH * EMB * 2);
    unsigned short* h_bf        = (unsigned short*)carve((size_t)LAYERS * BATCH * HID * 2);
    unsigned short* h_c_bf      = (unsigned short*)carve((size_t)BATCH * 2 * HID * 2);
    float* h_f32   = (float*)carve((size_t)LAYERS * BATCH * HID * 4);
    float* c_f32   = (float*)carve((size_t)LAYERS * BATCH * HID * 4);
    float* gates   = (float*)carve((size_t)BATCH * GATES * 4);
    float* scores  = (float*)carve((size_t)SRC * BATCH * 4);
    int*   tok     = (int*)carve((size_t)BATCH * 4);

    // one-time (per launch) weight conversion to bf16
    int n;
    n = GATES * EMB;
    f32_to_bf16_kernel<<<(n + 255) / 256, 256, 0, stream>>>(w_ih_l0, w_ih_l0_bf, n);
    n = GATES * HID;
    f32_to_bf16_kernel<<<(n + 255) / 256, 256, 0, stream>>>(w_hh_l0, w_hh_l0_bf, n);
    n = (LAYERS - 1) * GATES * HID;
    f32_to_bf16_kernel<<<(n + 255) / 256, 256, 0, stream>>>(w_ih_rest, w_ih_r_bf, n);
    f32_to_bf16_kernel<<<(n + 255) / 256, 256, 0, stream>>>(w_hh_rest, w_hh_r_bf, n);
    n = VOCAB * 2 * HID;
    f32_to_bf16_kernel<<<(n + 255) / 256, 256, 0, stream>>>(w_fc, w_fc_bf, n);

    init_state_kernel<<<(LAYERS * BATCH * HID + 255) / 256, 256, 0, stream>>>(
        h_f32, c_f32, h_bf, tok, start_code);

    for (int t = 0; t < TSTEPS; ++t) {
        embed_gather_kernel<<<(BATCH * EMB) / 256, 256, 0, stream>>>(emb, tok, x_bf);

        // layer 0: K1 = EMB
        lstm_gates_wmma_kernel<<<8, 256, 0, stream>>>(
            x_bf, EMB, h_bf, w_ih_l0_bf, w_hh_l0_bf, gates);
        lstm_pointwise_kernel<<<(BATCH * HID) / 256, 256, 0, stream>>>(
            gates, b_l0, h_f32, c_f32, h_bf);

        // layers 1..3: K1 = HID
        for (int l = 1; l < LAYERS; ++l) {
            lstm_gates_wmma_kernel<<<8, 256, 0, stream>>>(
                h_bf + (size_t)(l - 1) * BATCH * HID, HID,
                h_bf + (size_t)l * BATCH * HID,
                w_ih_r_bf + (size_t)(l - 1) * GATES * HID,
                w_hh_r_bf + (size_t)(l - 1) * GATES * HID,
                gates);
            lstm_pointwise_kernel<<<(BATCH * HID) / 256, 256, 0, stream>>>(
                gates, b_rest + (size_t)(l - 1) * GATES,
                h_f32 + (size_t)l * BATCH * HID,
                c_f32 + (size_t)l * BATCH * HID,
                h_bf + (size_t)l * BATCH * HID);
        }

        const float* h_top = h_f32 + (size_t)(LAYERS - 1) * BATCH * HID;
        attn_scores_kernel<<<SRC, BATCH, 0, stream>>>(enc_h, h_top, scores);
        attn_ctx_kernel<<<BATCH, SRC, 0, stream>>>(enc_h, h_top, scores, h_c_bf);

        float* out_t = out + (size_t)t * BATCH * VOCAB;
        fc_wmma_kernel<<<250, 256, 0, stream>>>(h_c_bf, w_fc_bf, b_fc, out_t);
        softmax_argmax_kernel<<<BATCH, 256, 0, stream>>>(out_t, tok);
    }
}
